// GraphEncoderDGL_420906795276
// MI455X (gfx1250) — compile-verified, matrix operand order checked by reading, not run on hardware
//
#include <hip/hip_runtime.h>

#define KNN_K 9

typedef __bf16 bf16_t;
typedef bf16_t v16bf __attribute__((ext_vector_type(16)));
typedef float  v8f   __attribute__((ext_vector_type(8)));
typedef unsigned short u16v16 __attribute__((ext_vector_type(16)));

__device__ __forceinline__ unsigned short f2bf(float f) {
  unsigned int u = __float_as_uint(f);
  u += 0x7FFFu + ((u >> 16) & 1u);          // round-to-nearest-even
  return (unsigned short)(u >> 16);
}

#define LDSTRIDE 40   // halves per LDS tile row: 16B-aligned rows, conflict padding

// ---------------------------------------------------------------------------
// GEMM + BN + activation + residual, bf16 WMMA (v_wmma_f32_16x16x32_bf16)
// Y[b] = act( BN( Wbf @ X[b] ) ) + R[b]
// Wbf: (M x K) bf16 row-major (pre-converted weights)
// XT : (B, N, K) bf16 n-major (pre-transposed activations)
// Y/R: (B, M, N) f32
// Requires M%64==0, K%32==0, N%64==0 (true for every GEMM in this network).
//
// Block = 128 threads = 4 waves. Workgroup tile = 64(M) x 64(N).
// Wave w owns M rows [w*16, w*16+16) and all four 16-wide N subtiles:
// one A fragment reused across 4 B fragments -> 4 back-to-back v_wmma/K-step.
// BOTH tiles staged with gfx1250 async DMA (global_load_async_to_lds_b128)
// into double-buffered LDS; next tile's copies overlap the current WMMAs
// (s_wait_asynccnt 4 exploits in-order async completion).
// ---------------------------------------------------------------------------
template <bool BN, bool RES, int ACT>
__global__ __launch_bounds__(128)
void gemm_bn_kernel(const unsigned short* __restrict__ Wbf,
                    const unsigned short* __restrict__ XT,
                    const float* __restrict__ gamma, const float* __restrict__ beta,
                    const float* __restrict__ resid, float* __restrict__ Y,
                    int M, int Kd, int N)
{
  __shared__ __align__(16) unsigned short lA[2][64 * LDSTRIDE];  // A tiles (M x K)
  __shared__ __align__(16) unsigned short lB[2][64 * LDSTRIDE];  // B tiles (N x K)

  const int b     = blockIdx.z;
  const int ntile = blockIdx.x * 64;
  const int mbase = blockIdx.y * 64;
  const int tid   = threadIdx.x;
  const int wave  = tid >> 5;
  const int lane  = tid & 31;
  const int g     = lane >> 4;   // lane group: selects K-half per ISA layout
  const int r     = lane & 15;

  const unsigned short* XTb = XT + (size_t)b * N * Kd;
  float*       Yb = Y + (size_t)b * M * N;
  const float* Rb = resid + (size_t)b * M * N;

  // async-copy assignment: 16B chunks; thread -> rows (tid>>2) and 32+(tid>>2),
  // segment (tid&3)*8 halves. 64 rows x 4 chunks = 256 chunks = 128 thr x 2.
  const int cRow = tid >> 2;
  const int cSeg = (tid & 3) * 8;
  const unsigned short* gA0 = Wbf + (size_t)(mbase + cRow) * Kd + cSeg;
  const unsigned short* gA1 = Wbf + (size_t)(mbase + 32 + cRow) * Kd + cSeg;
  const unsigned short* gB0 = XTb + (size_t)(ntile + cRow) * Kd + cSeg;
  const unsigned short* gB1 = XTb + (size_t)(ntile + 32 + cRow) * Kd + cSeg;

  unsigned ldsA0[2], ldsA1[2], ldsB0[2], ldsB1[2];
  #pragma unroll
  for (int bu = 0; bu < 2; ++bu) {
    ldsA0[bu] = (unsigned)(size_t)&lA[bu][cRow * LDSTRIDE + cSeg];
    ldsA1[bu] = (unsigned)(size_t)&lA[bu][(32 + cRow) * LDSTRIDE + cSeg];
    ldsB0[bu] = (unsigned)(size_t)&lB[bu][cRow * LDSTRIDE + cSeg];
    ldsB1[bu] = (unsigned)(size_t)&lB[bu][(32 + cRow) * LDSTRIDE + cSeg];
  }

  auto stage = [&](int k0, int bu) {
    unsigned long long a0 = (unsigned long long)(gA0 + k0);
    unsigned long long a1 = (unsigned long long)(gA1 + k0);
    unsigned long long b0 = (unsigned long long)(gB0 + k0);
    unsigned long long b1 = (unsigned long long)(gB1 + k0);
    asm volatile("global_load_async_to_lds_b128 %0, %1, off" :: "v"(ldsA0[bu]), "v"(a0) : "memory");
    asm volatile("global_load_async_to_lds_b128 %0, %1, off" :: "v"(ldsA1[bu]), "v"(a1) : "memory");
    asm volatile("global_load_async_to_lds_b128 %0, %1, off" :: "v"(ldsB0[bu]), "v"(b0) : "memory");
    asm volatile("global_load_async_to_lds_b128 %0, %1, off" :: "v"(ldsB1[bu]), "v"(b1) : "memory");
  };

  v8f acc[4];
  #pragma unroll
  for (int t = 0; t < 4; ++t) acc[t] = v8f{0.f,0.f,0.f,0.f,0.f,0.f,0.f,0.f};

  stage(0, 0);
  int ibuf = 0;
  for (int k0 = 0; k0 < Kd; k0 += 32, ibuf ^= 1) {
    // issue next tile's DMA into the other buffer, then wait only for the
    // current tile (async completion is in-order: cnt<=4 -> first 4 retired)
    if (k0 + 32 < Kd) {
      stage(k0 + 32, ibuf ^ 1);
      asm volatile("s_wait_asynccnt 0x4" ::: "memory");
    } else {
      asm volatile("s_wait_asynccnt 0x0" ::: "memory");
    }
    __syncthreads();

    const unsigned short* pA = lA[ibuf];
    const unsigned short* pB = lB[ibuf];

    // A fragment (ISA 16-bit A layout: K(j) = (j&7) + 8g + (j>>3)*16)
    u16v16 au;
    const int mrow = wave * 16 + r;
    #pragma unroll
    for (int v = 0; v < 8; ++v) {
      int kp = 2 * (v & 3) + 8 * g + (v >> 2) * 16;
      au[2 * v]     = pA[mrow * LDSTRIDE + kp];
      au[2 * v + 1] = pA[mrow * LDSTRIDE + kp + 1];
    }
    v16bf av = __builtin_bit_cast(v16bf, au);

    // gather ALL 4 B fragments, then 4 back-to-back WMMAs
    v16bf bv[4];
    #pragma unroll
    for (int t = 0; t < 4; ++t) {
      u16v16 bu;
      #pragma unroll
      for (int v = 0; v < 8; ++v) {
        int kp = 2 * (v & 3) + 8 * g + (v >> 2) * 16;
        bu[2 * v]     = pB[(16 * t + r) * LDSTRIDE + kp];
        bu[2 * v + 1] = pB[(16 * t + r) * LDSTRIDE + kp + 1];
      }
      bv[t] = __builtin_bit_cast(v16bf, bu);
    }
    #pragma unroll
    for (int t = 0; t < 4; ++t)
      acc[t] = __builtin_amdgcn_wmma_f32_16x16x32_bf16(false, av, false, bv[t],
                                                       (short)0, acc[t], false, false);
    __syncthreads();
  }

  // epilogue: D element i -> (M = i + 8g, N = r); straight-line per template
  #pragma unroll
  for (int t = 0; t < 4; ++t) {
    const int ncol = ntile + 16 * t + r;
    #pragma unroll
    for (int i = 0; i < 8; ++i) {
      int mg = mbase + wave * 16 + i + 8 * g;
      float y = acc[t][i];
      if (BN) y = y * gamma[mg] + beta[mg];
      if (ACT == 1) y = fmaxf(y, 0.f);
      if (RES) y += Rb[(size_t)mg * N + ncol];
      Yb[(size_t)mg * N + ncol] = y;
    }
  }
}

// ---------------------------------------------------------------------------
// Transpose + convert activations once per GEMM: X (B,K,N) f32 -> XT (B,N,K) bf16.
// LDS-tiled 32x32 so both global reads and writes are coalesced.
// grid (N/32, K/32, B), block (32, 8)
// ---------------------------------------------------------------------------
__global__ __launch_bounds__(256)
void tconv_kernel(const float* __restrict__ X, unsigned short* __restrict__ XT,
                  int Kd, int N)
{
  __shared__ unsigned short tile[32][34];
  const int n0 = blockIdx.x * 32;
  const int k0 = blockIdx.y * 32;
  const int b  = blockIdx.z;
  const float* Xb = X + (size_t)b * Kd * N;
  unsigned short* Tb = XT + (size_t)b * N * Kd;
  const int tx = threadIdx.x, ty = threadIdx.y;
  #pragma unroll
  for (int j = 0; j < 4; ++j) {
    int k = k0 + ty + 8 * j;
    tile[ty + 8 * j][tx] = f2bf(Xb[(size_t)k * N + n0 + tx]);
  }
  __syncthreads();
  #pragma unroll
  for (int j = 0; j < 4; ++j) {
    int n = n0 + ty + 8 * j;
    Tb[(size_t)n * Kd + k0 + tx] = tile[tx][ty + 8 * j];
  }
}

// ---------------------------------------------------------------------------
// Weight conversion: f32 -> bf16 (once per GEMM)
// ---------------------------------------------------------------------------
__global__ __launch_bounds__(256)
void w2bf_kernel(const float* __restrict__ in, unsigned short* __restrict__ out, int total)
{
  int t = blockIdx.x * 256 + threadIdx.x;
  if (t >= total) return;
  out[t] = f2bf(in[t]);
}

// ---------------------------------------------------------------------------
// Stem: y = leaky_relu( BN( stem_w(64x3) @ x(B,3,2048) ), 0.2 )
// ---------------------------------------------------------------------------
__global__ __launch_bounds__(256)
void stem_kernel(const float* __restrict__ x, const float* __restrict__ w,
                 const float* __restrict__ g, const float* __restrict__ be,
                 float* __restrict__ y, int total)
{
  int t = blockIdx.x * 256 + threadIdx.x;
  if (t >= total) return;
  int n = t & 2047;
  int o = (t >> 11) & 63;
  int b = t >> 17;
  const float* xb = x + (size_t)b * 3 * 2048 + n;
  float a = w[o*3+0] * xb[0] + w[o*3+1] * xb[2048] + w[o*3+2] * xb[4096];
  a = a * g[o] + be[o];
  y[t] = (a > 0.f) ? a : 0.2f * a;
}

// ---------------------------------------------------------------------------
// Downsample: 3x3 stride-(2,2) conv on (B,C,N,1); only kw==1 taps are ever
// in-bounds, so it is a strided 1-D conv (kernel 3) over N, then BN.
// ---------------------------------------------------------------------------
__global__ __launch_bounds__(256)
void down_kernel(const float* __restrict__ X, const float* __restrict__ Wd,
                 const float* __restrict__ g, const float* __restrict__ be,
                 float* __restrict__ Y, int Cin, int Nin, int Cout, int Nout, int total)
{
  int t = blockIdx.x * 256 + threadIdx.x;
  if (t >= total) return;
  int m = t % Nout;
  int o = (t / Nout) % Cout;
  int b = t / (Nout * Cout);
  const float* Xb = X + (size_t)b * Cin * Nin;
  int p0 = 2 * m - 1;
  float acc = 0.f;
  for (int ci = 0; ci < Cin; ++ci) {
    const float* w3  = Wd + (size_t)(o * Cin + ci) * 9 + 1;  // [kh*3 + 1]
    const float* row = Xb + (size_t)ci * Nin;
    float x0 = (p0 >= 0) ? row[p0] : 0.f;
    float x1 = row[p0 + 1];
    float x2 = (p0 + 2 < Nin) ? row[p0 + 2] : 0.f;
    acc += w3[0] * x0 + w3[3] * x1 + w3[6] * x2;
  }
  Y[t] = acc * g[o] + be[o];
}

// ---------------------------------------------------------------------------
// Squared norms: sq[b][n] = sum_c h[b][c][n]^2
// ---------------------------------------------------------------------------
__global__ __launch_bounds__(256)
void sq_kernel(const float* __restrict__ h, float* __restrict__ sq,
               int C, int N, int total)
{
  int t = blockIdx.x * 256 + threadIdx.x;
  if (t >= total) return;
  int n = t % N;
  int b = t / N;
  const float* hb = h + (size_t)b * C * N + n;
  float s = 0.f;
  for (int c = 0; c < C; ++c) { float v = hb[(size_t)c * N]; s += v * v; }
  sq[t] = s;
}

// ---------------------------------------------------------------------------
// kNN: for each (b,n), 9 smallest d = sq[n]+sq[m]-2<x_n,x_m>.
// Register-resident sorted list (append + unrolled bubble pass, statically
// indexed). Strict '<' with ascending m reproduces jax top_k tie order.
// ---------------------------------------------------------------------------
__global__ __launch_bounds__(128)
void knn_kernel(const float* __restrict__ h, const float* __restrict__ sq,
                int* __restrict__ idx, int C, int N)
{
  int n = blockIdx.x * 128 + threadIdx.x;
  int b = blockIdx.y;
  if (n >= N) return;
  const float* hb  = h + (size_t)b * C * N;
  const float* sqb = sq + (size_t)b * N;
  float bd[KNN_K];
  int   bi[KNN_K];
  #pragma unroll
  for (int i = 0; i < KNN_K; ++i) { bd[i] = 3.4e38f; bi[i] = 0; }
  float sn = sqb[n];

  for (int m0 = 0; m0 < N; m0 += 4) {
    float d0 = 0.f, d1 = 0.f, d2 = 0.f, d3 = 0.f;
    for (int c = 0; c < C; ++c) {
      const float* row = hb + (size_t)c * N;
      float xn = row[n];
      d0 += xn * row[m0];
      d1 += xn * row[m0 + 1];
      d2 += xn * row[m0 + 2];
      d3 += xn * row[m0 + 3];
    }
    float dd[4];
    dd[0] = sn + sqb[m0]     - 2.f * d0;
    dd[1] = sn + sqb[m0 + 1] - 2.f * d1;
    dd[2] = sn + sqb[m0 + 2] - 2.f * d2;
    dd[3] = sn + sqb[m0 + 3] - 2.f * d3;
    #pragma unroll
    for (int j = 0; j < 4; ++j) {
      float d = dd[j];
      if (d < bd[KNN_K - 1]) { bd[KNN_K - 1] = d; bi[KNN_K - 1] = m0 + j; }
      #pragma unroll
      for (int i = KNN_K - 1; i > 0; --i) {
        bool sw = bd[i] < bd[i - 1];
        float td = sw ? bd[i] : bd[i - 1];
        bd[i] = sw ? bd[i - 1] : bd[i];
        bd[i - 1] = td;
        int ti = sw ? bi[i] : bi[i - 1];
        bi[i] = sw ? bi[i - 1] : bi[i];
        bi[i - 1] = ti;
      }
    }
  }
  int* out = idx + ((size_t)b * N + n) * KNN_K;
  #pragma unroll
  for (int i = 0; i < KNN_K; ++i) out[i] = bi[i];
}

// ---------------------------------------------------------------------------
// Build stacked edge weights, directly in bf16:
// Wc(4C x C) = [ W2 ; W1 - W2 ], edge_w = [W1 | W2].
// ---------------------------------------------------------------------------
__global__ __launch_bounds__(256)
void edge_comb_kernel(const float* __restrict__ ew, unsigned short* __restrict__ wc,
                      int C, int total)
{
  int t = blockIdx.x * 256 + threadIdx.x;
  if (t >= total) return;
  int j = t % C;
  int o = t / C;
  int C2 = 2 * C;
  float v;
  if (o < C2) {
    v = ew[(size_t)o * C2 + C + j];                                   // W2
  } else {
    int rr = o - C2;
    v = ew[(size_t)rr * C2 + j] - ew[(size_t)rr * C2 + C + j];        // W1 - W2
  }
  wc[t] = f2bf(v);
}

// ---------------------------------------------------------------------------
// Edge gather-max epilogue:
// E[b,o,n] = relu( max_k ( gamma[o]*(Q[o,n] + P[o, idx[n,k]]) + beta[o] ) )
// PQ: (B, 4c, N) with P = rows [0,2c), Q = rows [2c,4c).
// grid = (N/256, 2c, B)
// ---------------------------------------------------------------------------
__global__ __launch_bounds__(256)
void edge_max_kernel(const float* __restrict__ PQ, const int* __restrict__ idx,
                     const float* __restrict__ gamma, const float* __restrict__ beta,
                     float* __restrict__ E, int C2, int N)
{
  int n = blockIdx.x * 256 + threadIdx.x;
  int o = blockIdx.y;
  int b = blockIdx.z;
  const float* Pb = PQ + (size_t)b * 2 * C2 * N;
  const float* Qb = Pb + (size_t)C2 * N;
  const float* Po = Pb + (size_t)o * N;
  float ga = gamma[o];
  float base = ga * Qb[(size_t)o * N + n] + beta[o];
  const int* id = idx + ((size_t)b * N + n) * KNN_K;
  float mx = -3.4e38f;
  #pragma unroll
  for (int k = 0; k < KNN_K; ++k) {
    float z = fmaf(ga, Po[id[k]], base);
    mx = fmaxf(mx, z);
  }
  E[((size_t)b * C2 + o) * N + n] = fmaxf(mx, 0.f);
}

// ---------------------------------------------------------------------------
// Mean over last stage (N=256, C=512):  mean[b][c]
// ---------------------------------------------------------------------------
__global__ __launch_bounds__(256)
void mean_kernel(const float* __restrict__ h, float* __restrict__ mean, int total)
{
  int t = blockIdx.x * 256 + threadIdx.x;
  if (t >= total) return;
  int c = t % 512;
  int b = t / 512;
  const float* hb = h + ((size_t)b * 512 + c) * 256;
  float s = 0.f;
  for (int n = 0; n < 256; ++n) s += hb[n];
  mean[t] = s * (1.f / 256.f);
}

// ---------------------------------------------------------------------------
// Final projection: out[b][e] = proj_w[e,:] @ mean[b,:] + proj_b[e]
// ---------------------------------------------------------------------------
__global__ __launch_bounds__(256)
void proj_kernel(const float* __restrict__ w, const float* __restrict__ pb,
                 const float* __restrict__ mean, float* __restrict__ out, int total)
{
  int t = blockIdx.x * 256 + threadIdx.x;
  if (t >= total) return;
  int e = t & 1023;
  int b = t >> 10;
  const float* wr = w + (size_t)e * 512;
  const float* mb = mean + (size_t)b * 512;
  float s = pb[e];
  for (int c = 0; c < 512; ++c) s += wr[c] * mb[c];
  out[t] = s;
}

// ---------------------------------------------------------------------------
// Host orchestration
// ---------------------------------------------------------------------------
extern "C" void kernel_launch(void* const* d_in, const int* in_sizes, int n_in,
                              void* d_out, int out_size, void* d_ws, size_t ws_size,
                              hipStream_t stream)
{
  (void)in_sizes; (void)n_in; (void)out_size; (void)ws_size;
  const int CH[4] = {64, 128, 256, 512};
  const int NB[4] = {2, 2, 6, 2};
  const int NS[4] = {2048, 1024, 512, 256};
  const int Bsz = 8;

  auto in = [&](int i) { return (const float*)d_in[i]; };

  // workspace carve-up (floats)
  float* w = (float*)d_ws;
  float* H0   = w; w += 1u << 20;      // h ping   (B*C*N == 1M floats every stage)
  float* H1   = w; w += 1u << 20;      // h pong
  float* T    = w; w += 1u << 22;      // t / ffn1 out (up to B*4c*n = 4M)
  float* PQ   = w; w += 1u << 22;      // stacked edge GEMM out (B*4c*n)
  float* E    = w; w += 1u << 21;      // edge-conv out (B*2c*n)
  unsigned short* WB  = (unsigned short*)w; w += 1u << 19;  // bf16 weights (<=1M halves)
  unsigned short* XTB = (unsigned short*)w; w += 1u << 21;  // bf16 transposed acts (<=4M halves)
  float* SQ   = w; w += 16384;         // squared norms (B*N)
  float* MEAN = w; w += 4096;          // (B,512)
  int*   IDX  = (int*)w;               // kNN indices (B*N*K)

  auto cvt = [&](const float* src, int total) {
    w2bf_kernel<<<(total + 255) / 256, 256, 0, stream>>>(src, WB, total);
  };
  auto tconv = [&](const float* X, int Kd, int N) {
    tconv_kernel<<<dim3(N / 32, Kd / 32, Bsz), dim3(32, 8), 0, stream>>>(X, XTB, Kd, N);
  };

  // template dispatch: (BN, RES, ACT)
  auto gemm = [&](const float* ga, const float* be,
                  const float* R, float* Y, int M, int Kd, int N, int act) {
    dim3 grid(N / 64, M / 64, Bsz);
    dim3 blk(128);
    if (!ga)
      gemm_bn_kernel<false, false, 0><<<grid, blk, 0, stream>>>(WB, XTB, ga, be, R, Y, M, Kd, N);
    else if (R)
      gemm_bn_kernel<true, true, 0><<<grid, blk, 0, stream>>>(WB, XTB, ga, be, R, Y, M, Kd, N);
    else if (act == 1)
      gemm_bn_kernel<true, false, 1><<<grid, blk, 0, stream>>>(WB, XTB, ga, be, R, Y, M, Kd, N);
    else
      gemm_bn_kernel<true, false, 0><<<grid, blk, 0, stream>>>(WB, XTB, ga, be, R, Y, M, Kd, N);
  };

  // inputs in setup_inputs() declaration order:
  // 0:x 1:stem_w 2:stem_g 3:stem_b 4:proj_w 5:proj_b then stages
  const float* x = in(0);
  int p = 6;

  float* h  = H0;
  float* hn = H1;

  // stem
  {
    int total = Bsz * 64 * 2048;
    stem_kernel<<<(total + 255) / 256, 256, 0, stream>>>(x, in(1), in(2), in(3), h, total);
  }

  for (int s = 0; s < 4; ++s) {
    int C = CH[s], N = NS[s];

    if (s > 0) {
      const float* dw = in(p++); const float* dg = in(p++); const float* db = in(p++);
      int Cin = CH[s - 1], Nin = NS[s - 1];
      int total = Bsz * C * N;
      down_kernel<<<(total + 255) / 256, 256, 0, stream>>>(h, dw, dg, db, hn,
                                                           Cin, Nin, C, N, total);
      float* tmp = h; h = hn; hn = tmp;
    }

    // kNN graph for this stage
    {
      int total = Bsz * N;
      sq_kernel<<<(total + 255) / 256, 256, 0, stream>>>(h, SQ, C, N, total);
      knn_kernel<<<dim3((N + 127) / 128, Bsz), 128, 0, stream>>>(h, SQ, IDX, C, N);
    }

    for (int blk = 0; blk < NB[s]; ++blk) {
      const float* fc1_w = in(p++); const float* fc1_g = in(p++); const float* fc1_b = in(p++);
      const float* ed_w  = in(p++); const float* ed_g  = in(p++); const float* ed_b  = in(p++);
      const float* fc2_w = in(p++); const float* fc2_g = in(p++); const float* fc2_b = in(p++);
      const float* f1_w  = in(p++); const float* f1_g  = in(p++); const float* f1_b  = in(p++);
      const float* f2_w  = in(p++); const float* f2_g  = in(p++); const float* f2_b  = in(p++);

      // t = BN(fc1 @ h)
      cvt(fc1_w, C * C);
      tconv(h, C, N);
      gemm(fc1_g, fc1_b, nullptr, T, C, C, N, 0);

      // stacked edge GEMM: PQ = [W2; W1-W2] @ t   (weights emitted in bf16)
      {
        int total = 4 * C * C;
        edge_comb_kernel<<<(total + 255) / 256, 256, 0, stream>>>(ed_w, WB, C, total);
      }
      tconv(T, C, N);
      gemm(nullptr, nullptr, nullptr, PQ, 4 * C, C, N, 0);

      // gather-max over K neighbors, BN+relu fused
      edge_max_kernel<<<dim3(N / 256, 2 * C, Bsz), 256, 0, stream>>>(
          PQ, IDX, ed_g, ed_b, E, 2 * C, N);

      // h = BN(fc2 @ E) + h
      cvt(fc2_w, C * 2 * C);
      tconv(E, 2 * C, N);
      gemm(fc2_g, fc2_b, h, hn, C, 2 * C, N, 0);
      { float* tmp = h; h = hn; hn = tmp; }

      // t = relu(BN(ffn1 @ h))
      cvt(f1_w, 4 * C * C);
      tconv(h, C, N);
      gemm(f1_g, f1_b, nullptr, T, 4 * C, C, N, 1);

      // h = BN(ffn2 @ t) + h
      cvt(f2_w, 4 * C * C);
      tconv(T, 4 * C, N);
      gemm(f2_g, f2_b, h, hn, C, 4 * C, N, 0);
      { float* tmp = h; h = hn; hn = tmp; }
    }
  }

  // mean over points, then projection
  {
    int total = Bsz * 512;
    mean_kernel<<<(total + 255) / 256, 256, 0, stream>>>(h, MEAN, total);
  }
  {
    int total = Bsz * 1024;
    proj_kernel<<<(total + 255) / 256, 256, 0, stream>>>(in(4), in(5), MEAN,
                                                         (float*)d_out, total);
  }
}
